// CrossAttentionBlock_82875688943654
// MI455X (gfx1250) — matrix operand order repaired and checked by reading, not compiled, not run on hardware
//
#include <hip/hip_runtime.h>
#include <hip/hip_bf16.h>

typedef __attribute__((ext_vector_type(16))) _Float16 v16h;
typedef __attribute__((ext_vector_type(8)))  float    v8f;

#define B_   2
#define T_   4096
#define C_   512
#define H_   8
#define DH_  64
#define HD_  512                      // H*DH
#define SCALE_ 0.35355339059327373f   // (DH//H)^-0.5 = 8^-0.5

static __device__ __forceinline__ v8f wmma_f16(v16h a, v16h b, v8f c) {
    return __builtin_amdgcn_wmma_f32_16x16x32_f16(false, a, false, b, (short)0, c, false, false);
}

// A-operand 16x32 f16 from f16 source (LDS or global), row-major, ld halves.
// lane<16: row=lane,    K = {0..15} (+k0)
// lane>=16: row=lane-16, K = {16..31} (+k0)
static __device__ __forceinline__ v16h load_a_f16(const _Float16* src, int ld, int row0, int k0) {
    int l  = threadIdx.x & 31;
    int r  = l & 15;
    int kb = k0 + ((l & 16) ? 16 : 0);
    const _Float16* p = src + (long)(row0 + r) * ld + kb;
    v16h a;
#pragma unroll
    for (int j = 0; j < 8; ++j) a[j]     = p[j];
#pragma unroll
    for (int j = 0; j < 8; ++j) a[8 + j] = p[8 + j];
    return a;
}

// B-operand 32x16: source stored K-contiguous per column (ld halves per column row).
// lane<16: col=lane, K=0..15 ; lane>=16: col=lane-16, K=16..31. 16 contiguous halves.
static __device__ __forceinline__ v16h load_b_kmaj(const _Float16* colbase, int ld, int n, int kb) {
    return *(const v16h*)(colbase + (long)n * ld + kb);
}

// ---------------- Kernel 0: weight convert + transpose  W[K][N] f32 -> Wt[N][K] f16 ----------
__global__ void __launch_bounds__(256) ca_wt(const float* __restrict__ w,
                                             _Float16* __restrict__ wt, int N) {
    int idx = blockIdx.x * 256 + threadIdx.x;    // grid sized exactly K*N/256, K=512
    int k = idx & (C_ - 1);
    int n = idx >> 9;
    wt[idx] = (_Float16)w[(long)k * N + n];
}

// ------------- Macro-tile GEMM: block = 256 thr (8 waves), tile M=128 x N=64, K-step 32 ------
// A staged f32->f16 in LDS [128][32]; B staged from Wt (f16, [N][K]) in LDS [64][32].

// ---------------- Kernel 1: QK projection ----------------
// grid 1024: mt = bid>>4 (64), nt = bid&15 (16 col groups: 0..7 Q heads, 8..15 K heads)
__global__ void __launch_bounds__(256) ca_proj_qk(const float* __restrict__ cond,
                                                  const _Float16* __restrict__ wt_qk,
                                                  _Float16* __restrict__ qw,
                                                  _Float16* __restrict__ kw) {
    __shared__ __align__(32) _Float16 a_lds[128 * 32];
    __shared__ __align__(32) _Float16 b_lds[64 * 32];

    int mt = blockIdx.x >> 4, nt = blockIdx.x & 15;
    int row0 = mt * 128, n0 = nt * 64;
    int wave = threadIdx.x >> 5, lane = threadIdx.x & 31;
    int tid = threadIdx.x;

    v8f acc[4] = {};
    for (int k0 = 0; k0 < C_; k0 += 32) {
        __syncthreads();
        {   // A: 128 rows x 32 halves (convert f32->f16), 16 elems/thread
            int r = tid >> 1, ko = (tid & 1) * 16;
            const float* p = cond + (long)(row0 + r) * C_ + k0 + ko;
            _Float16* q = a_lds + r * 32 + ko;
#pragma unroll
            for (int j = 0; j < 16; ++j) q[j] = (_Float16)p[j];
        }
        {   // B: 64 cols x 32 halves from Wt rows (contiguous), 8 halves/thread
            int n = tid >> 2, c = (tid & 3) * 8;
            *(uint4*)(b_lds + n * 32 + c) =
                *(const uint4*)(wt_qk + (long)(n0 + n) * C_ + k0 + c);
        }
        __syncthreads();
        v16h a = load_a_f16(a_lds, 32, wave * 16, 0);
        int kb = (lane & 16) ? 16 : 0, nn = lane & 15;
#pragma unroll
        for (int g = 0; g < 4; ++g) {
            v16h b = load_b_kmaj(b_lds, 32, g * 16 + nn, kb);
            acc[g] = wmma_f16(a, b, acc[g]);
        }
    }
    bool isQ = (nt < 8);
    int h = nt & 7;
    float scale = isQ ? SCALE_ : 1.0f;
    _Float16* dst = isQ ? qw : kw;
    int row0w = row0 + wave * 16;
    int b  = row0w >> 12;
    int t0 = row0w & (T_ - 1);
    _Float16* base = dst + (((long)b * H_ + h) * T_ + t0) * DH_;
    int rb = (lane & 16) ? 8 : 0, nn = lane & 15;
#pragma unroll
    for (int g = 0; g < 4; ++g)
#pragma unroll
        for (int j = 0; j < 8; ++j)
            base[(long)(rb + j) * DH_ + g * 16 + nn] = (_Float16)(acc[g][j] * scale);
}

// ---------------- Kernel 2: V projection -> V^T layout [B,H,DH,T] ----------------
// grid 512: mt = bid>>3 (64), h = bid&7
__global__ void __launch_bounds__(256) ca_proj_v(const float* __restrict__ x,
                                                 const _Float16* __restrict__ wt_v,
                                                 _Float16* __restrict__ vt) {
    __shared__ __align__(32) _Float16 a_lds[128 * 32];
    __shared__ __align__(32) _Float16 b_lds[64 * 32];

    int mt = blockIdx.x >> 3, h = blockIdx.x & 7;
    int row0 = mt * 128, n0 = h * 64;
    int wave = threadIdx.x >> 5, lane = threadIdx.x & 31;
    int tid = threadIdx.x;

    v8f acc[4] = {};
    for (int k0 = 0; k0 < C_; k0 += 32) {
        __syncthreads();
        {
            int r = tid >> 1, ko = (tid & 1) * 16;
            const float* p = x + (long)(row0 + r) * C_ + k0 + ko;
            _Float16* q = a_lds + r * 32 + ko;
#pragma unroll
            for (int j = 0; j < 16; ++j) q[j] = (_Float16)p[j];
        }
        {
            int n = tid >> 2, c = (tid & 3) * 8;
            *(uint4*)(b_lds + n * 32 + c) =
                *(const uint4*)(wt_v + (long)(n0 + n) * C_ + k0 + c);
        }
        __syncthreads();
        v16h a = load_a_f16(a_lds, 32, wave * 16, 0);
        int kb = (lane & 16) ? 16 : 0, nn = lane & 15;
#pragma unroll
        for (int g = 0; g < 4; ++g) {
            v16h b = load_b_kmaj(b_lds, 32, g * 16 + nn, kb);
            acc[g] = wmma_f16(a, b, acc[g]);
        }
    }
    int row0w = row0 + wave * 16;
    int b  = row0w >> 12;
    int t0 = row0w & (T_ - 1);
    int rb = (lane & 16) ? 8 : 0, nn = lane & 15;
    _Float16* base = vt + ((long)b * H_ + h) * DH_ * T_;
#pragma unroll
    for (int g = 0; g < 4; ++g)
#pragma unroll
        for (int j = 0; j < 8; ++j)
            base[(long)(g * 16 + nn) * T_ + t0 + rb + j] = (_Float16)acc[g][j];
}

// ---------------- Kernel 3: flash attention, 64-key chunks, double-buffered LDS -------------
// grid = B*H*(T/128) = 512 blocks x 256 threads (8 waves, 16 queries each).
__global__ void __launch_bounds__(256) ca_flash(const _Float16* __restrict__ qw,
                                                const _Float16* __restrict__ kw,
                                                const _Float16* __restrict__ vt,
                                                _Float16* __restrict__ ow) {
    __shared__ __align__(32) _Float16 kbuf[2][64 * 64];   // [key][dh]
    __shared__ __align__(32) _Float16 vbuf[2][64 * 64];   // [dh][key]
    __shared__ __align__(32) _Float16 pbuf[8 * 16 * 64];  // per-wave [q][key]

    int bh   = blockIdx.x >> 5;
    int qblk = blockIdx.x & 31;
    int b    = bh >> 3;
    int h    = bh & 7;
    int wave = threadIdx.x >> 5;
    int lane = threadIdx.x & 31;
    int tid  = threadIdx.x;
    int q0   = qblk * 128 + wave * 16;

    const _Float16* qbase  = qw + (((long)b * H_ + h) * T_) * DH_;
    const _Float16* kbase  = kw + (((long)b * H_ + h) * T_) * DH_;
    const _Float16* vtbase = vt + ((long)b * H_ + h) * DH_ * T_;

    v16h a0 = load_a_f16(qbase, DH_, q0, 0);
    v16h a1 = load_a_f16(qbase, DH_, q0, 32);

    v8f o[4] = {};
    float m[8], lsum[8];
#pragma unroll
    for (int j = 0; j < 8; ++j) { m[j] = -1e30f; lsum[j] = 0.0f; }

    _Float16* pb = pbuf + wave * (16 * 64);
    int prow = (lane & 16) ? 8 : 0;
    int pcol = lane & 15;
    int kb   = (lane & 16) ? 16 : 0;
    int nn   = lane & 15;

    // per-thread staging geometry
    int vd0 = tid >> 3,         voff0 = (tid & 7) * 8;          // chunk tid
    int vd1 = (tid + 256) >> 3, voff1 = ((tid + 256) & 7) * 8;  // chunk tid+256

    // prologue: stage chunk 0 into buffer 0
    {
        const uint4* src = (const uint4*)kbase;                 // 64x64 halves contiguous
        ((uint4*)kbuf[0])[tid]       = src[tid];
        ((uint4*)kbuf[0])[tid + 256] = src[tid + 256];
        ((uint4*)vbuf[0])[tid]       = *(const uint4*)(vtbase + (long)vd0 * T_ + voff0);
        ((uint4*)vbuf[0])[tid + 256] = *(const uint4*)(vtbase + (long)vd1 * T_ + voff1);
    }
    __syncthreads();

    const int NCH = T_ / 64;
    for (int i = 0; i < NCH; ++i) {
        int cur = i & 1, nxt = cur ^ 1;
        const _Float16* kc = kbuf[cur];
        const _Float16* vc = vbuf[cur];

        // issue next chunk's global loads early (latency hidden behind compute)
        uint4 nk0, nk1, nv0, nv1;
        bool have_next = (i + 1) < NCH;
        if (have_next) {
            long j0n = (long)(i + 1) * 64;
            const uint4* src = (const uint4*)(kbase + j0n * DH_);
            nk0 = src[tid];
            nk1 = src[tid + 256];
            nv0 = *(const uint4*)(vtbase + (long)vd0 * T_ + j0n + voff0);
            nv1 = *(const uint4*)(vtbase + (long)vd1 * T_ + j0n + voff1);
            if (i + 2 < NCH)   // warm L2 one more chunk ahead (global_prefetch_b8)
                __builtin_prefetch(kbase + (long)(i + 2) * 64 * DH_ + tid * 16, 0, 0);
        }

        // S = Q K^T : four 16x16 key tiles, contraction over dh=64 (2 k-steps)
        v8f s[4] = {};
#pragma unroll
        for (int t = 0; t < 4; ++t)
            s[t] = wmma_f16(a0, *(const v16h*)&kc[(t * 16 + nn) * 64 + 0 + kb], s[t]);
#pragma unroll
        for (int t = 0; t < 4; ++t)
            s[t] = wmma_f16(a1, *(const v16h*)&kc[(t * 16 + nn) * 64 + 32 + kb], s[t]);

        // online softmax (row j -> VGPR j / j+8; cols -> lanes within half)
#pragma unroll
        for (int j = 0; j < 8; ++j) {
            float mx = fmaxf(fmaxf(s[0][j], s[1][j]), fmaxf(s[2][j], s[3][j]));
            mx = fmaxf(mx, __shfl_xor(mx, 1, 32));
            mx = fmaxf(mx, __shfl_xor(mx, 2, 32));
            mx = fmaxf(mx, __shfl_xor(mx, 4, 32));
            mx = fmaxf(mx, __shfl_xor(mx, 8, 32));
            float mnew = fmaxf(m[j], mx);
            float e0 = __expf(s[0][j] - mnew);
            float e1 = __expf(s[1][j] - mnew);
            float e2 = __expf(s[2][j] - mnew);
            float e3 = __expf(s[3][j] - mnew);
            s[0][j] = e0; s[1][j] = e1; s[2][j] = e2; s[3][j] = e3;
            float sm = (e0 + e1) + (e2 + e3);
            sm += __shfl_xor(sm, 1, 32);
            sm += __shfl_xor(sm, 2, 32);
            sm += __shfl_xor(sm, 4, 32);
            sm += __shfl_xor(sm, 8, 32);
            float rs = __expf(m[j] - mnew);
            lsum[j] = lsum[j] * rs + sm;
            m[j] = mnew;
            o[0][j] *= rs; o[1][j] *= rs; o[2][j] *= rs; o[3][j] *= rs;
        }

        // P (C/D layout) -> per-wave LDS [16][64], reload as two A operands
#pragma unroll
        for (int j = 0; j < 8; ++j)
#pragma unroll
            for (int t = 0; t < 4; ++t)
                pb[(prow + j) * 64 + t * 16 + pcol] = (_Float16)s[t][j];
        v16h pa0 = load_a_f16(pb, 64, 0, 0);
        v16h pa1 = load_a_f16(pb, 64, 0, 32);

        // O += P @ V : B from vbuf[dh][key], 2 k-steps x 4 dh col-groups
#pragma unroll
        for (int g = 0; g < 4; ++g)
            o[g] = wmma_f16(pa0, *(const v16h*)&vc[(g * 16 + nn) * 64 + 0 + kb], o[g]);
#pragma unroll
        for (int g = 0; g < 4; ++g)
            o[g] = wmma_f16(pa1, *(const v16h*)&vc[(g * 16 + nn) * 64 + 32 + kb], o[g]);

        // commit next chunk into the other buffer
        if (have_next) {
            ((uint4*)kbuf[nxt])[tid]       = nk0;
            ((uint4*)kbuf[nxt])[tid + 256] = nk1;
            ((uint4*)vbuf[nxt])[tid]       = nv0;
            ((uint4*)vbuf[nxt])[tid + 256] = nv1;
        }
        __syncthreads();
    }

    // normalize and write attn output in [B, T, H*DH] (f16)
    _Float16* obase = ow + ((long)b * T_) * HD_ + (long)h * DH_;
    int rb = (lane & 16) ? 8 : 0;
#pragma unroll
    for (int j = 0; j < 8; ++j) {
        float inv = 1.0f / lsum[j];
        int t = q0 + rb + j;
#pragma unroll
        for (int g = 0; g < 4; ++g)
            obase[(long)t * HD_ + g * 16 + nn] = (_Float16)(o[g][j] * inv);
    }
}

// ---------------- Kernel 4: output projection attn @ W_u + b_u -> f32 ----------------
// grid 512: mt = bid>>3 (64), cg = bid&7
__global__ void __launch_bounds__(256) ca_out_proj(const _Float16* __restrict__ aw,
                                                   const _Float16* __restrict__ wt_u,
                                                   const float* __restrict__ b_u,
                                                   float* __restrict__ out) {
    __shared__ __align__(32) _Float16 a_lds[128 * 32];
    __shared__ __align__(32) _Float16 b_lds[64 * 32];

    int mt = blockIdx.x >> 3, cg = blockIdx.x & 7;
    int row0 = mt * 128, n0 = cg * 64;
    int wave = threadIdx.x >> 5, lane = threadIdx.x & 31;
    int tid = threadIdx.x;

    v8f acc[4] = {};
    for (int k0 = 0; k0 < HD_; k0 += 32) {
        __syncthreads();
        {   // A: 128 rows x 32 halves from f16 activations, 16B copies
#pragma unroll
            for (int i = 0; i < 2; ++i) {
                int c = tid + i * 256;                 // 512 chunks
                int r = c >> 2, off = (c & 3) * 8;
                *((uint4*)a_lds + c) =
                    *(const uint4*)(aw + (long)(row0 + r) * HD_ + k0 + off);
            }
        }
        {
            int n = tid >> 2, c = (tid & 3) * 8;
            *(uint4*)(b_lds + n * 32 + c) =
                *(const uint4*)(wt_u + (long)(n0 + n) * HD_ + k0 + c);
        }
        __syncthreads();
        v16h a = load_a_f16(a_lds, 32, wave * 16, 0);
        int kb = (lane & 16) ? 16 : 0, nn = lane & 15;
#pragma unroll
        for (int g = 0; g < 4; ++g) {
            v16h b = load_b_kmaj(b_lds, 32, g * 16 + nn, kb);
            acc[g] = wmma_f16(a, b, acc[g]);
        }
    }
    int row0w = row0 + wave * 16;
    int rb = (lane & 16) ? 8 : 0, nn = lane & 15;
#pragma unroll
    for (int g = 0; g < 4; ++g) {
        int col = n0 + g * 16 + nn;
        float bias = b_u[col];
#pragma unroll
        for (int j = 0; j < 8; ++j)
            out[(long)(row0w + rb + j) * C_ + col] = acc[g][j] + bias;
    }
}

extern "C" void kernel_launch(void* const* d_in, const int* in_sizes, int n_in,
                              void* d_out, int out_size, void* d_ws, size_t ws_size,
                              hipStream_t stream) {
    (void)in_sizes; (void)n_in; (void)out_size; (void)ws_size;
    const float* x    = (const float*)d_in[0];
    const float* cond = (const float*)d_in[1];
    const float* w_qk = (const float*)d_in[2];
    const float* w_v  = (const float*)d_in[3];
    const float* w_u  = (const float*)d_in[4];
    const float* b_u  = (const float*)d_in[5];
    float* out = (float*)d_out;

    char* ws = (char*)d_ws;
    _Float16* qw    = (_Float16*)(ws);                          // 8 MB  [B,H,T,DH]
    _Float16* kw    = (_Float16*)(ws + ((size_t)8  << 20));     // 8 MB  [B,H,T,DH]
    _Float16* vt    = (_Float16*)(ws + ((size_t)16 << 20));     // 8 MB  [B,H,DH,T]
    _Float16* aw    = (_Float16*)(ws + ((size_t)24 << 20));     // 8 MB  [B,T,H*DH]
    _Float16* wt_qk = (_Float16*)(ws + ((size_t)32 << 20));     // 1 MB  [1024][512]
    _Float16* wt_v  = (_Float16*)(ws + ((size_t)33 << 20));     // 512KB [512][512]
    _Float16* wt_u  = (_Float16*)(ws + ((size_t)33 << 20) + ((size_t)512 << 10));

    ca_wt<<<2048, 256, 0, stream>>>(w_qk, wt_qk, 2 * HD_);      // 1024x512
    ca_wt<<<1024, 256, 0, stream>>>(w_v, wt_v, HD_);            // 512x512
    ca_wt<<<1024, 256, 0, stream>>>(w_u, wt_u, C_);             // 512x512

    ca_proj_qk<<<1024, 256, 0, stream>>>(cond, wt_qk, qw, kw);
    ca_proj_v <<<512,  256, 0, stream>>>(x, wt_v, vt);
    ca_flash  <<<512,  256, 0, stream>>>(qw, kw, vt, aw);
    ca_out_proj<<<512, 256, 0, stream>>>(aw, wt_u, b_u, out);
}
